// MultiSelfAttention_28535762715292
// MI455X (gfx1250) — matrix operand order repaired and checked by reading, not compiled, not run on hardware
//
#include <hip/hip_runtime.h>
#include <hip/hip_bf16.h>

// ---------- types ----------
typedef __attribute__((ext_vector_type(16))) __bf16 v16bf;
typedef __attribute__((ext_vector_type(8)))  float  v8f;
typedef __attribute__((ext_vector_type(4)))  unsigned int u32x4; // trivial 16B vector

union Frag {            // one WMMA 16-bit A/B fragment: 16 bf16 = 8 VGPRs
  v16bf v;
  u32x4 q[2];
};

#define WMMA_BF16(a, b, c) \
  __builtin_amdgcn_wmma_f32_16x16x32_bf16(false, (a), false, (b), (short)0, (c), false, false)

// async global->LDS copy of 16 bytes (per lane); tracked by ASYNCcnt
#define ASYNC_CP_B128(lds_u32, gptr)                                        \
  asm volatile("global_load_async_to_lds_b128 %0, %1, off"                  \
               :: "v"(lds_u32), "v"((unsigned long long)(gptr)) : "memory")

static __device__ __forceinline__ unsigned lds_addr(const void* p) {
  // LDS aperture: flat addr[31:0] == LDS byte offset (ISA 10.2)
  return (unsigned)(unsigned long long)p;
}

// ---------- fp32 -> bf16 conversion ----------
__global__ void cvt_f32_to_bf16(const float* __restrict__ in, __bf16* __restrict__ out, int n) {
  int i = blockIdx.x * blockDim.x + threadIdx.x;
  int stride = gridDim.x * blockDim.x;
  for (; i < n; i += stride) out[i] = (__bf16)in[i];
}

// ---------- tiled WMMA GEMM:  Co[M,N] = (A[M,K] @ W[N,K]^T + bias) * alpha ----------
// 128x128 block tile, KT=64 stages, double-buffered LDS filled by async-to-LDS copies.
template <typename OutT>
__global__ __launch_bounds__(256) void gemm_bf16_wmma(
    const __bf16* __restrict__ A,    // [M,K] row-major, bf16
    const __bf16* __restrict__ W,    // [N,K] row-major, bf16  (B = W^T)
    const float*  __restrict__ bias, // [N] fp32
    OutT* __restrict__ Co,           // [M,N]
    int M, int N, int K, float alpha)
{
  constexpr int KT  = 64;                 // K per LDS stage (2 WMMA k-steps)
  constexpr int LDA = 72;                 // padded LDS row stride (144B, 16B aligned)
  __shared__ __attribute__((aligned(16))) __bf16 As[2][128 * LDA];
  __shared__ __attribute__((aligned(16))) __bf16 Bs[2][128 * LDA];

  const int tid  = threadIdx.x;
  const int lane = tid & 31;
  const int w    = tid >> 5;              // 8 waves
  const int wm   = (w >> 2) * 64;         // wave M offset within block tile
  const int wn   = (w & 3) * 32;          // wave N offset within block tile
  const int m0   = blockIdx.y * 128;
  const int n0   = blockIdx.x * 128;

  v8f acc[4][2];
#pragma unroll
  for (int mt = 0; mt < 4; ++mt)
#pragma unroll
    for (int nt = 0; nt < 2; ++nt)
#pragma unroll
      for (int r = 0; r < 8; ++r) acc[mt][nt][r] = 0.0f;

  // cooperative stage loads: 2 threads per row, 32 bf16 (64B = 4 async b128) each
  const int lrow = tid >> 1;              // 0..127
  const int lcol = (tid & 1) * 32;        // 0 or 32
  const __bf16* Ag = A + (size_t)(m0 + lrow) * K + lcol;
  const __bf16* Bg = W + (size_t)(n0 + lrow) * K + lcol;
  const unsigned laA0 = lds_addr(&As[0][lrow * LDA + lcol]);
  const unsigned laB0 = lds_addr(&Bs[0][lrow * LDA + lcol]);
  const unsigned bufBytes = (unsigned)(128 * LDA * 2);

  // issue one stage's async copies (8 per thread, completes in order)
  auto issue_stage = [&](int buf, int kt) {
    const __bf16* ga = Ag + kt;
    const __bf16* gb = Bg + kt;
    const unsigned la = laA0 + (unsigned)buf * bufBytes;
    const unsigned lb = laB0 + (unsigned)buf * bufBytes;
#pragma unroll
    for (int i = 0; i < 4; ++i) {
      ASYNC_CP_B128(la + i * 16, ga + i * 8);
      ASYNC_CP_B128(lb + i * 16, gb + i * 8);
    }
  };

  const int nstages = K / KT;
  issue_stage(0, 0);

  for (int st = 0; st < nstages; ++st) {
    const int buf = st & 1;
    const bool has_next = (st + 1) < nstages;
    if (has_next) issue_stage(buf ^ 1, (st + 1) * KT);

    // wait for *this* stage's 8 copies (next stage's 8 may stay in flight)
    if (has_next) asm volatile("s_wait_asynccnt 0x8" ::: "memory");
    else          asm volatile("s_wait_asynccnt 0x0" ::: "memory");
    __syncthreads();                      // all waves' stage data visible in LDS

#pragma unroll
    for (int kk = 0; kk < 2; ++kk) {      // 2 WMMA k-steps per stage
      // A fragments: lane holds row (lane&15); chunks at 8*(lane>>4) and +16
      Frag af[4];
      const int c0 = (lane >> 4) * 8;
#pragma unroll
      for (int mt = 0; mt < 4; ++mt) {
        const __bf16* p = &As[buf][(wm + mt * 16 + (lane & 15)) * LDA + kk * 32];
        af[mt].q[0] = *(const u32x4*)(p + c0);
        af[mt].q[1] = *(const u32x4*)(p + 16 + c0);
      }
      // B fragments: lane holds col (lane&15); chunks at 8*(lane>>4) and +16
#pragma unroll
      for (int nt = 0; nt < 2; ++nt) {
        Frag bfr;
        const __bf16* p = &Bs[buf][(wn + nt * 16 + (lane & 15)) * LDA + kk * 32 + (lane >> 4) * 8];
        bfr.q[0] = *(const u32x4*)(p);
        bfr.q[1] = *(const u32x4*)(p + 16);
#pragma unroll
        for (int mt = 0; mt < 4; ++mt)
          acc[mt][nt] = WMMA_BF16(af[mt].v, bfr.v, acc[mt][nt]);
      }
    }
    __syncthreads();                      // done reading buf before it is refilled
  }

  // epilogue: C/D layout -> n = lane&15, m = r + 8*(lane>>4)
#pragma unroll
  for (int mt = 0; mt < 4; ++mt)
#pragma unroll
    for (int nt = 0; nt < 2; ++nt) {
      const int n  = n0 + wn + nt * 16 + (lane & 15);
      const int mb = m0 + wm + mt * 16 + ((lane >> 4) << 3);
      const float bv = bias[n];
#pragma unroll
      for (int r = 0; r < 8; ++r) {
        float val = (acc[mt][nt][r] + bv) * alpha;
        Co[(size_t)(mb + r) * N + n] = (OutT)val;
      }
    }
}

// ---------- causal flash attention (bf16 WMMA, fp32 softmax) ----------
// Q/K/V/Y: [B, T, H*64] bf16, head h lives at column offset h*64. Q pre-scaled by 1/sqrt(d).
__global__ __launch_bounds__(128) void flash_attn_wmma(
    const __bf16* __restrict__ Qm, const __bf16* __restrict__ Km,
    const __bf16* __restrict__ Vm, __bf16* __restrict__ Ym, int T)
{
  constexpr int Cd = 1024, D = 64, H = 16;
  constexpr int LDK = 72;                  // 144B row stride, 16B aligned
  __shared__ __attribute__((aligned(16))) __bf16 Ks[64 * LDK]; // K tile: row = key, col = d
  __shared__ __attribute__((aligned(16))) __bf16 Vt[64 * LDK]; // V tile transposed: row = d, col = key
  __shared__ __attribute__((aligned(16))) __bf16 Ps[64 * LDK]; // P staging for layout transpose

  const int tid  = threadIdx.x;
  const int lane = tid & 31;
  const int w    = tid >> 5;               // 4 waves, each owns 16 q rows
  const int q0   = blockIdx.x * 64;
  const int b    = blockIdx.y / H;
  const int h    = blockIdx.y % H;
  const size_t base = ((size_t)b * T) * Cd + (size_t)h * D;

  // Q fragments for this wave's rows (row = q0 + w*16 + (lane&15)), K-dim = d (2 chunks of 32)
  Frag qf[2];
  {
    const __bf16* qp = Qm + base + (size_t)(q0 + w * 16 + (lane & 15)) * Cd;
    const int c0 = (lane >> 4) * 8;
#pragma unroll
    for (int kc = 0; kc < 2; ++kc) {
      qf[kc].q[0] = *(const u32x4*)(qp + kc * 32 + c0);
      qf[kc].q[1] = *(const u32x4*)(qp + kc * 32 + 16 + c0);
    }
  }

  v8f o[4];
#pragma unroll
  for (int nt = 0; nt < 4; ++nt)
#pragma unroll
    for (int r = 0; r < 8; ++r) o[nt][r] = 0.0f;
  float mrow[8], lsum[8];
#pragma unroll
  for (int r = 0; r < 8; ++r) { mrow[r] = -1e30f; lsum[r] = 0.0f; }

  const int jrow = tid >> 1;               // 0..63 : key row this thread loads
  const int jcol = (tid & 1) * 32;         // d-range half

  for (int k0 = 0; k0 <= q0; k0 += 64) {
    // --- K tile via async-to-LDS; V tile via register transpose ---
    {
      const __bf16* kp = Km + base + (size_t)(k0 + jrow) * Cd + jcol;
      const unsigned ks = lds_addr(Ks + jrow * LDK + jcol);
#pragma unroll
      for (int i = 0; i < 4; ++i) ASYNC_CP_B128(ks + i * 16, kp + i * 8);

      const __bf16* vp = Vm + base + (size_t)(k0 + jrow) * Cd + jcol;
      union { u32x4 q[4]; __bf16 e[32]; } vv;
#pragma unroll
      for (int i = 0; i < 4; ++i) vv.q[i] = *(const u32x4*)(vp + i * 8);
#pragma unroll
      for (int c = 0; c < 32; ++c) Vt[(jcol + c) * LDK + jrow] = vv.e[c];

      asm volatile("s_wait_asynccnt 0x0" ::: "memory");
    }
    __syncthreads();

    // --- S = Q @ K^T (16 x 64 per wave) ---
    v8f s[4];
#pragma unroll
    for (int nt = 0; nt < 4; ++nt) {
      v8f z;
#pragma unroll
      for (int r = 0; r < 8; ++r) z[r] = 0.0f;
#pragma unroll
      for (int kc = 0; kc < 2; ++kc) {
        Frag bfr;
        const __bf16* p = Ks + (nt * 16 + (lane & 15)) * LDK + kc * 32 + (lane >> 4) * 8;
        bfr.q[0] = *(const u32x4*)(p);
        bfr.q[1] = *(const u32x4*)(p + 16);
        z = WMMA_BF16(qf[kc].v, bfr.v, z);
      }
      s[nt] = z;
    }

    // --- causal mask (only the diagonal tile needs it; T % 64 == 0) ---
    const int mloc = (lane >> 4) << 3;
    const int ncol = lane & 15;
    if (k0 == q0) {
#pragma unroll
      for (int nt = 0; nt < 4; ++nt)
#pragma unroll
        for (int r = 0; r < 8; ++r)
          if (nt * 16 + ncol > w * 16 + mloc + r) s[nt][r] = -1e30f;
    }

    // --- online softmax: row reductions across the 16-lane C/D group ---
#pragma unroll
    for (int r = 0; r < 8; ++r) {
      float mx = s[0][r];
#pragma unroll
      for (int nt = 1; nt < 4; ++nt) mx = fmaxf(mx, s[nt][r]);
#pragma unroll
      for (int off = 1; off < 16; off <<= 1) mx = fmaxf(mx, __shfl_xor(mx, off, 32));
      const float mnew = fmaxf(mrow[r], mx);
      const float corr = __expf(mrow[r] - mnew);
      mrow[r] = mnew;
      float rs = 0.0f;
#pragma unroll
      for (int nt = 0; nt < 4; ++nt) {
        float p = __expf(s[nt][r] - mnew);
        s[nt][r] = p;
        rs += p;
      }
#pragma unroll
      for (int off = 1; off < 16; off <<= 1) rs += __shfl_xor(rs, off, 32);
      lsum[r] = lsum[r] * corr + rs;
#pragma unroll
      for (int nt = 0; nt < 4; ++nt) o[nt][r] *= corr;
    }

    // --- P: C/D layout -> A layout via LDS (wave-private region) ---
#pragma unroll
    for (int nt = 0; nt < 4; ++nt)
#pragma unroll
      for (int r = 0; r < 8; ++r)
        Ps[(w * 16 + mloc + r) * LDK + nt * 16 + ncol] = (__bf16)s[nt][r];
    asm volatile("s_wait_dscnt 0x0" ::: "memory");  // wave-local LDS RAW

    Frag pf[2];
    {
      const __bf16* pp = Ps + (w * 16 + (lane & 15)) * LDK;
      const int c0 = (lane >> 4) * 8;
#pragma unroll
      for (int kc = 0; kc < 2; ++kc) {
        pf[kc].q[0] = *(const u32x4*)(pp + kc * 32 + c0);
        pf[kc].q[1] = *(const u32x4*)(pp + kc * 32 + 16 + c0);
      }
    }

    // --- O += P @ V  (B-fragments contiguous thanks to transposed Vt) ---
#pragma unroll
    for (int nt = 0; nt < 4; ++nt)
#pragma unroll
      for (int kc = 0; kc < 2; ++kc) {
        Frag bfr;
        const __bf16* p = Vt + (nt * 16 + (lane & 15)) * LDK + kc * 32 + (lane >> 4) * 8;
        bfr.q[0] = *(const u32x4*)(p);
        bfr.q[1] = *(const u32x4*)(p + 16);
        o[nt] = WMMA_BF16(pf[kc].v, bfr.v, o[nt]);
      }
    __syncthreads();
  }

  // --- epilogue: y = O / l ---
  const int mloc = (lane >> 4) << 3;
#pragma unroll
  for (int nt = 0; nt < 4; ++nt) {
    const int dd = nt * 16 + (lane & 15);
#pragma unroll
    for (int r = 0; r < 8; ++r) {
      const int row = q0 + w * 16 + mloc + r;
      Ym[base + (size_t)row * Cd + dd] = (__bf16)(o[nt][r] / lsum[r]);
    }
  }
}

// ---------- host-side orchestration ----------
extern "C" void kernel_launch(void* const* d_in, const int* in_sizes, int n_in,
                              void* d_out, int out_size, void* d_ws, size_t ws_size,
                              hipStream_t stream) {
  (void)in_sizes; (void)n_in; (void)out_size; (void)ws_size;
  const float* x  = (const float*)d_in[0];
  const float* Wq = (const float*)d_in[1];
  const float* bq = (const float*)d_in[2];
  const float* Wk = (const float*)d_in[3];
  const float* bk = (const float*)d_in[4];
  const float* Wv = (const float*)d_in[5];
  const float* bv = (const float*)d_in[6];
  const float* Wp = (const float*)d_in[7];
  const float* bp = (const float*)d_in[8];
  float* out = (float*)d_out;

  const int B = 4, T = 2048, C = 1024, H = 16;
  const int M = B * T;                        // 8192

  // workspace layout (bf16): x | Wq | Wk | Wv | Wp | q | k | v | y  (~92 MB)
  __bf16* xb  = (__bf16*)d_ws;
  __bf16* wqb = xb  + (size_t)M * C;
  __bf16* wkb = wqb + (size_t)C * C;
  __bf16* wvb = wkb + (size_t)C * C;
  __bf16* wpb = wvb + (size_t)C * C;
  __bf16* qb  = wpb + (size_t)C * C;
  __bf16* kb  = qb  + (size_t)M * C;
  __bf16* vb  = kb  + (size_t)M * C;
  __bf16* yb  = vb  + (size_t)M * C;

  cvt_f32_to_bf16<<<4096, 256, 0, stream>>>(x,  xb,  M * C);
  cvt_f32_to_bf16<<<1024, 256, 0, stream>>>(Wq, wqb, C * C);
  cvt_f32_to_bf16<<<1024, 256, 0, stream>>>(Wk, wkb, C * C);
  cvt_f32_to_bf16<<<1024, 256, 0, stream>>>(Wv, wvb, C * C);
  cvt_f32_to_bf16<<<1024, 256, 0, stream>>>(Wp, wpb, C * C);

  dim3 gg(C / 128, M / 128);                  // (8, 64)
  // fold attention scale 1/sqrt(64) into Q
  gemm_bf16_wmma<__bf16><<<gg, 256, 0, stream>>>(xb, wqb, bq, qb, M, C, C, 0.125f);
  gemm_bf16_wmma<__bf16><<<gg, 256, 0, stream>>>(xb, wkb, bk, kb, M, C, C, 1.0f);
  gemm_bf16_wmma<__bf16><<<gg, 256, 0, stream>>>(xb, wvb, bv, vb, M, C, C, 1.0f);

  dim3 ga(T / 64, B * H);                     // (32, 64)
  flash_attn_wmma<<<ga, 128, 0, stream>>>(qb, kb, vb, yb, T);

  gemm_bf16_wmma<float><<<gg, 256, 0, stream>>>(yb, wpb, bp, out, M, C, C, 1.0f);
}
